// Net_10101763080662
// MI455X (gfx1250) — matrix operand order repaired and checked by reading, not compiled, not run on hardware
//
#include <hip/hip_runtime.h>
#include <hip/hip_bf16.h>
#include <stdint.h>

// ---------------------------------------------------------------------------
// GNN pipeline for gfx1250 (MI455X): KNN -> EdgeConv MLP (WMMA f16) ->
// 3-way aggregation -> nn1/nn2 (WMMA f16) -> segment reductions -> 2-layer GRU
// -> head. Big GEMMs use v_wmma_f32_16x16x32_f16 (f32 accumulate).
// Workspace requirement: ~263 MB (see offsets below).
// ---------------------------------------------------------------------------

typedef __attribute__((ext_vector_type(16))) _Float16 v16h;
typedef __attribute__((ext_vector_type(8)))  _Float16 v8h;
typedef __attribute__((ext_vector_type(8)))  float    v8f;

#define N_PER 128
#define NUM_GRAPHS 2048
#define NTOT (N_PER * NUM_GRAPHS)     // 262144 nodes
#define KNN 4
#define NEDGE (NTOT * KNN)            // 1048576 edges
#define BATCH 1024
#define GRUH 32

// workspace byte offsets (all 256B aligned)
static constexpr unsigned long long OFF_WFRAG = 0ull;                       // 104 frags * 512 halves
static constexpr unsigned long long OFF_NBR   = 106496ull;                  // NEDGE int32
static constexpr unsigned long long OFF_H1    = OFF_NBR + 4ull * NEDGE;     // NTOT*384 f16
static constexpr unsigned long long OFF_H2    = OFF_H1 + 2ull * NTOT * 384ull; // NTOT*64 f32
static constexpr unsigned long long OFF_G     = OFF_H2 + 4ull * NTOT * 64ull;  // 2048*256 f32
static constexpr unsigned long long OFF_HB0   = OFF_G + 4ull * NUM_GRAPHS * 256ull;
static constexpr unsigned long long OFF_HB1   = OFF_HB0 + 4ull * BATCH * GRUH;
static constexpr unsigned long long OFF_O0    = OFF_HB1 + 4ull * BATCH * GRUH;
static constexpr unsigned long long OFF_O1    = OFF_O0 + 4ull * BATCH * GRUH; // 2*1024*32 f32

__device__ __forceinline__ v8f wmma16(v16h a, v16h b, v8f c) {
  // D = A(16x32 f16) * B(32x16 f16) + C(16x16 f32)
  return __builtin_amdgcn_wmma_f32_16x16x32_f16(false, a, false, b, (short)0, c,
                                                false, false);
}

// ---------------------------------------------------------------------------
// K0: pre-swizzle all weights into WMMA B-fragment order.
// frag layout: frag f -> 32 lanes x 16 halves at wfrag[f*512 + lane*16 + e].
// B-fragment element (lane, e): K = (e<8?0:16)+(lane<16?0:8)+(e&7) (+32*kstep), N = tile*16+lane%16.
// frags 0..3   : W1   [10x64]  padded K->32            (edge MLP layer 1)
// frags 4..19  : W2   [64x128] f = 4 + tile*2 + kstep  (edge MLP layer 2)
// frags 20..91 : nn1_W[384x94] padded N->96, f = 20 + kstep*6 + tile
// frags 92..103: nn2_W[94x64]  padded K->96, f = 92 + kstep*4 + tile
// ---------------------------------------------------------------------------
__global__ void k_prep(const float* __restrict__ W1, const float* __restrict__ W2,
                       const float* __restrict__ nn1W, const float* __restrict__ nn2W,
                       _Float16* __restrict__ wfrag) {
  int gid = blockIdx.x * blockDim.x + threadIdx.x;
  if (gid >= 104 * 512) return;
  int f = gid >> 9, r = gid & 511, lane = r >> 4, e = r & 15;
  int hi = (lane < 16) ? 0 : 8;
  int ke = ((e < 8) ? 0 : 16) + hi + (e & 7);
  int c16 = lane & 15;
  float v = 0.f;
  if (f < 4) {
    int k = ke, n = f * 16 + c16;
    if (k < 10) v = W1[k * 64 + n];
  } else if (f < 20) {
    int idx = f - 4, tile = idx >> 1, ks = idx & 1;
    int k = ks * 32 + ke, n = tile * 16 + c16;
    v = W2[k * 128 + n];
  } else if (f < 92) {
    int idx = f - 20, ks = idx / 6, tile = idx % 6;
    int k = ks * 32 + ke, n = tile * 16 + c16;
    if (n < 94) v = nn1W[k * 94 + n];
  } else {
    int idx = f - 92, ks = idx >> 2, tile = idx & 3;
    int k = ks * 32 + ke, n = tile * 16 + c16;
    if (k < 94) v = nn2W[k * 64 + n];
  }
  wfrag[gid] = (_Float16)v;
}

// ---------------------------------------------------------------------------
// K1: per-graph KNN(4) on x[:, :3]; graphs are contiguous 128-node blocks.
// ---------------------------------------------------------------------------
__global__ void k_knn(const float* __restrict__ x, int* __restrict__ nbr) {
  __shared__ float px[N_PER], py[N_PER], pz[N_PER];
  int i = threadIdx.x;
  int base = blockIdx.x * N_PER;
  px[i] = x[(base + i) * 5 + 0];
  py[i] = x[(base + i) * 5 + 1];
  pz[i] = x[(base + i) * 5 + 2];
  __syncthreads();
  float bd0 = 1e30f, bd1 = 1e30f, bd2 = 1e30f, bd3 = 1e30f;
  int bi0 = 0, bi1 = 0, bi2 = 0, bi3 = 0;
  float xi = px[i], yi = py[i], zi = pz[i];
  for (int j = 0; j < N_PER; j++) {
    if (j == i) continue;
    float dx = px[j] - xi, dy = py[j] - yi, dz = pz[j] - zi;
    float d = dx * dx + dy * dy + dz * dz;
    if (d < bd3) {
      if (d < bd2) {
        bd3 = bd2; bi3 = bi2;
        if (d < bd1) {
          bd2 = bd1; bi2 = bi1;
          if (d < bd0) { bd1 = bd0; bi1 = bi0; bd0 = d; bi0 = j; }
          else { bd1 = d; bi1 = j; }
        } else { bd2 = d; bi2 = j; }
      } else { bd3 = d; bi3 = j; }
    }
  }
  int o = (base + i) * 4;
  nbr[o + 0] = base + bi0; nbr[o + 1] = base + bi1;
  nbr[o + 2] = base + bi2; nbr[o + 3] = base + bi3;
}

// ---------------------------------------------------------------------------
// K2: edge MLP + 3-way aggregation. One wave per 16-edge tile (= 4 nodes).
// h1[node][0..127]=max, [128..255]=mean, [256..383]=sum (all >=0 post-ReLU,
// so the reference's outer ReLU is a no-op).
// ---------------------------------------------------------------------------
__global__ void __launch_bounds__(256) k_edge(
    const float* __restrict__ x, const int* __restrict__ nbr,
    const float* __restrict__ b1, const float* __restrict__ b2,
    const _Float16* __restrict__ wfrag, _Float16* __restrict__ h1) {
  __shared__ __align__(32) _Float16 ldsW2[16 * 512];   // 16KB: W2 B-fragments
  __shared__ __align__(32) _Float16 ldsY[8][16 * 64];  // 16KB: per-wave y1 tile
  int tid = threadIdx.x;
  {  // stage W2 fragments (frags 4..19) into LDS once per block
    const uint4* src = (const uint4*)(wfrag + 4 * 512);
    uint4* dst = (uint4*)ldsW2;
    for (int idx = tid; idx < (16 * 512 * 2) / 16; idx += 256) dst[idx] = src[idx];
  }
  __syncthreads();
  int w = tid >> 5, lane = tid & 31;
  int c16 = lane & 15;
  int base8 = (lane < 16) ? 0 : 8;
  v16h b1f[4];  // hoisted W1 B-fragments (K padded with zeros)
#pragma unroll
  for (int t = 0; t < 4; t++)
    b1f[t] = *(const v16h*)(wfrag + ((t * 32 + lane) << 4));
  _Float16* myY = ldsY[w];
  int waveG = blockIdx.x * 8 + w;
  int nwaves = gridDim.x * 8;
  for (int tile = waveG; tile < NEDGE / 16; tile += nwaves) {
    int edge = tile * 16 + c16;      // row M = lane%16
    int node = edge >> 2;
    int nj = nbr[edge];
    float f10[10];
#pragma unroll
    for (int c = 0; c < 5; c++) {
      float xi = x[node * 5 + c];
      f10[c] = xi;
      f10[c + 5] = x[nj * 5 + c] - xi;
    }
    v16h af;  // A fragment: K = (e<8?0:16) + base8 + (e&7), zero pad K>=10
#pragma unroll
    for (int e = 0; e < 16; e++) {
      int k = ((e < 8) ? 0 : 16) + base8 + (e & 7);
      af[e] = (k < 10) ? (_Float16)f10[k] : (_Float16)0.0f;
    }
    // layer 1: [16x10]x[10x64], bias + ReLU, restage into LDS as f16
#pragma unroll
    for (int t = 0; t < 4; t++) {
      v8f acc = {};
      acc = wmma16(af, b1f[t], acc);
      int n = t * 16 + c16;
      float bb = b1[n];
#pragma unroll
      for (int r = 0; r < 8; r++) {
        float m = fmaxf(acc[r] + bb, 0.f);
        myY[(r + base8) * 64 + n] = (_Float16)m;  // D row = r + base8
      }
    }
    // rebuild A fragments for layer 2 (per-wave DS ops are in-order)
    v16h a2[2];
#pragma unroll
    for (int s = 0; s < 2; s++) {
      const _Float16* ap = myY + c16 * 64 + s * 32 + base8;
      v8h lo = *(const v8h*)ap;
      v8h hv = *(const v8h*)(ap + 16);
#pragma unroll
      for (int e = 0; e < 8; e++) { a2[s][e] = lo[e]; a2[s][e + 8] = hv[e]; }
    }
    // layer 2: [16x64]x[64x128], bias + ReLU, then max/mean/sum over K=4.
    // D rows 0..3 / 4..7 (lanes<16) and 8..11 / 12..15 (lanes>=16) are whole
    // nodes, so aggregation is purely per-lane across acc VGPRs.
    int nodeBase = tile * 4;
#pragma unroll
    for (int t = 0; t < 8; t++) {
      v8f acc = {};
#pragma unroll
      for (int s = 0; s < 2; s++) {
        v16h bf = *(const v16h*)(ldsW2 + (((t * 2 + s) * 32 + lane) << 4));
        acc = wmma16(a2[s], bf, acc);
      }
      int n = t * 16 + c16;
      float bb = b2[n];
      float m[8];
#pragma unroll
      for (int r = 0; r < 8; r++) m[r] = fmaxf(acc[r] + bb, 0.f);
      float sA = m[0] + m[1] + m[2] + m[3];
      float xA = fmaxf(fmaxf(m[0], m[1]), fmaxf(m[2], m[3]));
      float sB = m[4] + m[5] + m[6] + m[7];
      float xB = fmaxf(fmaxf(m[4], m[5]), fmaxf(m[6], m[7]));
      int nA = nodeBase + ((lane < 16) ? 0 : 2);
      unsigned long long oA = (unsigned long long)nA * 384 + n;
      unsigned long long oB = oA + 384;  // node nA+1
      h1[oA] = (_Float16)xA; h1[oA + 128] = (_Float16)(sA * 0.25f); h1[oA + 256] = (_Float16)sA;
      h1[oB] = (_Float16)xB; h1[oB + 128] = (_Float16)(sB * 0.25f); h1[oB + 256] = (_Float16)sB;
    }
  }
}

// ---------------------------------------------------------------------------
// K3: h2 = relu(h1 @ nn1_W + b) @ nn2_W + b. One wave per 16-node tile.
// ---------------------------------------------------------------------------
__global__ void __launch_bounds__(256) k_mlp(
    const _Float16* __restrict__ h1, const float* __restrict__ nn1b,
    const float* __restrict__ nn2b, const _Float16* __restrict__ wfrag,
    float* __restrict__ h2) {
  __shared__ __align__(32) _Float16 ldsW4[12 * 512];   // 12KB nn2 fragments
  __shared__ __align__(32) _Float16 ldsT[8][16 * 96];  // 24KB per-wave mid tile
  int tid = threadIdx.x;
  {
    const uint4* src = (const uint4*)(wfrag + 92 * 512);
    uint4* dst = (uint4*)ldsW4;
    for (int idx = tid; idx < (12 * 512 * 2) / 16; idx += 256) dst[idx] = src[idx];
  }
  __syncthreads();
  int w = tid >> 5, lane = tid & 31;
  int c16 = lane & 15, base8 = (lane < 16) ? 0 : 8;
  _Float16* myT = ldsT[w];
  const _Float16* w3 = wfrag + 20 * 512;
  int waveG = blockIdx.x * 8 + w, nwaves = gridDim.x * 8;
  for (int tile = waveG; tile < NTOT / 16; tile += nwaves) {
    int rowBase = tile * 16;
    v8f acc[6] = {};
    for (int s = 0; s < 12; s++) {  // K = 384 -> 12 steps
      const _Float16* ap =
          h1 + (unsigned long long)(rowBase + c16) * 384 + s * 32 + base8;
      v8h lo = *(const v8h*)ap;
      v8h hv = *(const v8h*)(ap + 16);
      v16h a;
#pragma unroll
      for (int e = 0; e < 8; e++) { a[e] = lo[e]; a[e + 8] = hv[e]; }
#pragma unroll
      for (int t = 0; t < 6; t++) {  // N = 94 padded to 96
        v16h bf = *(const v16h*)(w3 + (((s * 6 + t) * 32 + lane) << 4));
        acc[t] = wmma16(a, bf, acc[t]);
      }
    }
#pragma unroll
    for (int t = 0; t < 6; t++) {  // bias + ReLU; zero the 2 pad columns
      int n = t * 16 + c16;
      float bb = (n < 94) ? nn1b[n] : 0.f;
#pragma unroll
      for (int r = 0; r < 8; r++) {
        float m = (n < 94) ? fmaxf(acc[t][r] + bb, 0.f) : 0.f;
        myT[(r + base8) * 96 + n] = (_Float16)m;
      }
    }
    v8f acc2[4] = {};
#pragma unroll
    for (int s = 0; s < 3; s++) {  // K = 94 padded to 96
      const _Float16* ap = myT + c16 * 96 + s * 32 + base8;
      v8h lo = *(const v8h*)ap;
      v8h hv = *(const v8h*)(ap + 16);
      v16h a;
#pragma unroll
      for (int e = 0; e < 8; e++) { a[e] = lo[e]; a[e + 8] = hv[e]; }
#pragma unroll
      for (int t = 0; t < 4; t++) {
        v16h bf = *(const v16h*)(ldsW4 + (((s * 4 + t) * 32 + lane) << 4));
        acc2[t] = wmma16(a, bf, acc2[t]);
      }
    }
#pragma unroll
    for (int t = 0; t < 4; t++) {
      int n = t * 16 + c16;
      float bb = nn2b[n];
#pragma unroll
      for (int r = 0; r < 8; r++)
        h2[(unsigned long long)(rowBase + r + base8) * 64 + n] = acc2[t][r] + bb;
    }
  }
}

// ---------------------------------------------------------------------------
// K4: per-graph segment max/min/sum/mean over 128 nodes, ReLU, g[2048,256].
// ---------------------------------------------------------------------------
__global__ void k_seg(const float* __restrict__ h2, float* __restrict__ g) {
  int gi = blockIdx.x, c = threadIdx.x;  // 64 threads = 64 cols
  const float* p = h2 + (unsigned long long)gi * N_PER * 64 + c;
  float mx = -1e30f, mn = 1e30f, sm = 0.f;
  for (int n = 0; n < N_PER; n++) {
    float v = p[n * 64];
    mx = fmaxf(mx, v); mn = fminf(mn, v); sm += v;
  }
  float* go = g + gi * 256;
  go[c]       = fmaxf(mx, 0.f);
  go[64 + c]  = fmaxf(mn, 0.f);
  go[128 + c] = fmaxf(sm, 0.f);
  go[192 + c] = fmaxf(sm * (1.f / 128.f), 0.f);
}

// ---------------------------------------------------------------------------
// K5a: GRU layer-0 step. seq_in[t][b][f] = g[t*1024 + f*4 + (b>>8)][b&255]
// (faithful to reshape(T,256,1024).transpose(0,2,1)). Block = 8 batch x 32 h.
// ---------------------------------------------------------------------------
__global__ void k_gru0(const float* __restrict__ g, const float* __restrict__ hprev,
                       const float* __restrict__ wih, const float* __restrict__ whh,
                       const float* __restrict__ bih, const float* __restrict__ bhh,
                       float* __restrict__ hout, float* __restrict__ xout, int t) {
  __shared__ float xsh[8][256];
  __shared__ float hsh[8][32];
  int tid = threadIdx.x;
  int bBase = blockIdx.x * 8;
  for (int idx = tid; idx < 8 * 256; idx += 256) {
    int bl = idx >> 8, f = idx & 255;
    int b = bBase + bl;
    xsh[bl][f] = g[(unsigned long long)(t * 1024 + f * 4 + (b >> 8)) * 256 + (b & 255)];
  }
  { int bl = tid >> 5, h = tid & 31; hsh[bl][h] = hprev[(bBase + bl) * 32 + h]; }
  __syncthreads();
  int bl = tid >> 5, h = tid & 31;
  int b = bBase + bl;
  float ir = bih[h], iz = bih[32 + h], inn = bih[64 + h];
  for (int f = 0; f < 256; f++) {
    float xv = xsh[bl][f];
    ir  += xv * wih[h * 256 + f];
    iz  += xv * wih[(32 + h) * 256 + f];
    inn += xv * wih[(64 + h) * 256 + f];
  }
  float hr = bhh[h], hz = bhh[32 + h], hn = bhh[64 + h];
#pragma unroll
  for (int j = 0; j < 32; j++) {
    float hv = hsh[bl][j];
    hr += hv * whh[h * 32 + j];
    hz += hv * whh[(32 + h) * 32 + j];
    hn += hv * whh[(64 + h) * 32 + j];
  }
  float r = 1.f / (1.f + expf(-(ir + hr)));
  float z = 1.f / (1.f + expf(-(iz + hz)));
  float n = tanhf(inn + r * hn);
  float hnew = (1.f - z) * n + z * hsh[bl][h];
  hout[b * 32 + h] = hnew;
  xout[b * 32 + h] = hnew;
}

// K5b: GRU layer-1 step (input dim 32); also writes out1[t].
__global__ void k_gru1(const float* __restrict__ xin, const float* __restrict__ hprev,
                       const float* __restrict__ wih, const float* __restrict__ whh,
                       const float* __restrict__ bih, const float* __restrict__ bhh,
                       float* __restrict__ hout, float* __restrict__ out1, int t) {
  __shared__ float xsh[8][32];
  __shared__ float hsh[8][32];
  int tid = threadIdx.x;
  int bBase = blockIdx.x * 8;
  int bl = tid >> 5, h = tid & 31;
  xsh[bl][h] = xin[(bBase + bl) * 32 + h];
  hsh[bl][h] = hprev[(bBase + bl) * 32 + h];
  __syncthreads();
  int b = bBase + bl;
  float ir = bih[h], iz = bih[32 + h], inn = bih[64 + h];
  float hr = bhh[h], hz = bhh[32 + h], hn = bhh[64 + h];
#pragma unroll
  for (int j = 0; j < 32; j++) {
    float xv = xsh[bl][j];
    ir  += xv * wih[h * 32 + j];
    iz  += xv * wih[(32 + h) * 32 + j];
    inn += xv * wih[(64 + h) * 32 + j];
    float hv = hsh[bl][j];
    hr += hv * whh[h * 32 + j];
    hz += hv * whh[(32 + h) * 32 + j];
    hn += hv * whh[(64 + h) * 32 + j];
  }
  float r = 1.f / (1.f + expf(-(ir + hr)));
  float z = 1.f / (1.f + expf(-(iz + hz)));
  float n = tanhf(inn + r * hn);
  float hnew = (1.f - z) * n + z * hsh[bl][h];
  hout[b * 32 + h] = hnew;
  out1[(t * 1024 + b) * 32 + h] = hnew;
}

// ---------------------------------------------------------------------------
// K6: y[r] = relu(relu(y_row) @ nn4_W + nn4_b) with the exact transpose/
// reshape scramble; then append hT = [h_layer0, h_layer1].
// ---------------------------------------------------------------------------
__global__ void k_final(const float* __restrict__ out1, const float* __restrict__ hb0,
                        const float* __restrict__ hb1, const float* __restrict__ nn4W,
                        const float* __restrict__ nn4b, float* __restrict__ out) {
  int i = blockIdx.x * blockDim.x + threadIdx.x;
  if (i < 2048) {
    int t = i >> 10, q = i & 1023;
    int hsel = q >> 5, bb = (q & 31) * 32;
    float acc = 0.f;
#pragma unroll
    for (int c = 0; c < 32; c++) {
      float y = out1[(unsigned long long)(t * 1024 + bb + c) * 32 + hsel];
      acc += fmaxf(y, 0.f) * nn4W[c];
    }
    out[i] = fmaxf(acc + nn4b[0], 0.f);
  } else if (i < 2048 + 2 * 1024 * 32) {
    int j = i - 2048;
    int l = j >> 15, rem = j & 32767;
    out[i] = (l == 0) ? hb0[rem] : hb1[rem];
  }
}

extern "C" void kernel_launch(void* const* d_in, const int* in_sizes, int n_in,
                              void* d_out, int out_size, void* d_ws, size_t ws_size,
                              hipStream_t stream) {
  const float* x    = (const float*)d_in[0];
  // d_in[1] = batch (unused: graphs are contiguous 128-node blocks)
  const float* h1in = (const float*)d_in[2];   // [2,1024,32]
  const float* W1   = (const float*)d_in[3];
  const float* b1   = (const float*)d_in[4];
  const float* W2   = (const float*)d_in[5];
  const float* b2   = (const float*)d_in[6];
  const float* nn1W = (const float*)d_in[7];
  const float* nn1b = (const float*)d_in[8];
  const float* nn2W = (const float*)d_in[9];
  const float* nn2b = (const float*)d_in[10];
  const float* nn4W = (const float*)d_in[11];
  const float* nn4b = (const float*)d_in[12];
  const float* wih0 = (const float*)d_in[13];
  const float* whh0 = (const float*)d_in[14];
  const float* bih0 = (const float*)d_in[15];
  const float* bhh0 = (const float*)d_in[16];
  const float* wih1 = (const float*)d_in[17];
  const float* whh1 = (const float*)d_in[18];
  const float* bih1 = (const float*)d_in[19];
  const float* bhh1 = (const float*)d_in[20];

  char* ws = (char*)d_ws;
  _Float16* wfrag = (_Float16*)(ws + OFF_WFRAG);
  int*      nbr   = (int*)(ws + OFF_NBR);
  _Float16* h1    = (_Float16*)(ws + OFF_H1);
  float*    h2    = (float*)(ws + OFF_H2);
  float*    g     = (float*)(ws + OFF_G);
  float*    hb0   = (float*)(ws + OFF_HB0);
  float*    hb1   = (float*)(ws + OFF_HB1);
  float*    o0    = (float*)(ws + OFF_O0);
  float*    o1    = (float*)(ws + OFF_O1);
  float*    out   = (float*)d_out;

  k_prep<<<208, 256, 0, stream>>>(W1, W2, nn1W, nn2W, wfrag);
  k_knn<<<NUM_GRAPHS, N_PER, 0, stream>>>(x, nbr);
  k_edge<<<1024, 256, 0, stream>>>(x, nbr, b1, b2, wfrag, h1);
  k_mlp<<<512, 256, 0, stream>>>(h1, nn1b, nn2b, wfrag, h2);
  k_seg<<<NUM_GRAPHS, 64, 0, stream>>>(h2, g);
  // GRU: T=2, layers sequenced via stream order
  k_gru0<<<128, 256, 0, stream>>>(g, h1in, wih0, whh0, bih0, bhh0, hb0, o0, 0);
  k_gru1<<<128, 256, 0, stream>>>(o0, h1in + 1024 * 32, wih1, whh1, bih1, bhh1, hb1, o1, 0);
  k_gru0<<<128, 256, 0, stream>>>(g, hb0, wih0, whh0, bih0, bhh0, hb0, o0, 1);
  k_gru1<<<128, 256, 0, stream>>>(o0, hb1, wih1, whh1, bih1, bhh1, hb1, o1, 1);
  k_final<<<264, 256, 0, stream>>>(o1, hb0, hb1, nn4W, nn4b, out);
}
